// DyHeadBlockWithDCNV3_43920335569436
// MI455X (gfx1250) — matrix operand order repaired and checked by reading, not compiled, not run on hardware
//
#include <hip/hip_runtime.h>
#include <math.h>

// ---------------------------------------------------------------------------
// DyHead block with DCNv3 for MI455X (gfx1250, wave32, WMMA).
// Offset/mask grouped conv = implicit GEMM on v_wmma_f32_16x16x32_f16:
//   - weights pre-packed to f16 [g][oc pad32][K=576] -> B frags are 2x b128
//   - A tile staged in LDS per 64-K super-chunk (== one conv tap), shared by
//     2 waves (both output-channel tiles); frags read as ds_load_b128
// All gather/normalize stages are bandwidth-bound VALU kernels.
// ---------------------------------------------------------------------------

typedef __attribute__((ext_vector_type(16))) _Float16 v16h;
typedef __attribute__((ext_vector_type(8)))  _Float16 v8h;
typedef __attribute__((ext_vector_type(8)))  float    v8f;

#define NB  4     // batch
#define CC  256   // channels
#define GG  4     // dcn groups
#define CGc 64    // channels per group
#define OCG 27    // offset-conv outputs per group (2*9 offsets + 9 mask)
#define OCP 32    // padded outputs per group
#define OMC 108   // total offset-conv output channels
#define KTOT 576  // 64 ch * 9 taps

static __device__ __forceinline__ float sigf(float x) { return 1.f / (1.f + expf(-x)); }

// ---------------- pack offset-conv weights to f16, [g][oc(32)][K=576] -------
__global__ void pack_weights_kernel(const float* __restrict__ ow, _Float16* __restrict__ wf) {
  int idx = blockIdx.x * blockDim.x + threadIdx.x;
  const int total = GG * OCP * KTOT;          // 73728
  if (idx >= total) return;
  int k  = idx % KTOT;
  int oc = (idx / KTOT) % OCP;
  int g  = idx / (KTOT * OCP);
  int tap = k >> 6, ci = k & 63;
  float v = (oc < OCG) ? ow[((size_t)(g * OCG + oc) * CGc + ci) * 9 + tap] : 0.f;
  wf[idx] = (_Float16)v;
}

// ---------------- depthwise 3x3 + scale/bias + SiLU -------------------------
__global__ void dw_silu_kernel(const float* __restrict__ x, const float* __restrict__ w,
                               const float* __restrict__ s, const float* __restrict__ b,
                               float* __restrict__ y, int H, int W) {
  int idx = blockIdx.x * blockDim.x + threadIdx.x;
  int total = NB * CC * H * W;
  if (idx >= total) return;
  int ox = idx % W, oy = (idx / W) % H;
  int c  = (idx / (H * W)) % CC;
  const float* xp = x + (size_t)(idx / (H * W)) * H * W;  // (n,c) plane
  float acc = 0.f;
#pragma unroll
  for (int ky = 0; ky < 3; ++ky)
#pragma unroll
    for (int kx = 0; kx < 3; ++kx) {
      int iy = oy + ky - 1, ix = ox + kx - 1;
      if (iy >= 0 && iy < H && ix >= 0 && ix < W)
        acc += xp[iy * W + ix] * w[c * 9 + ky * 3 + kx];
    }
  float t = acc * s[c] + b[c];
  y[idx] = t * sigf(t);
}

// ---------------- grouped offset/mask conv as WMMA implicit GEMM ------------
// grid: (ceil(HW/16), NB*GG), block: 64 (2 waves; wave w -> oc tile w)
__global__ __launch_bounds__(64)
void offset_conv_wmma_kernel(const float* __restrict__ mid, const _Float16* __restrict__ wf,
                             const float* __restrict__ ob, float* __restrict__ om,
                             int H, int W) {
  __shared__ _Float16 Alds[16 * 64];          // 16 pixels x 64 K (one tap)  = 2KB
  const int HW   = H * W;
  const int tid  = threadIdx.x;               // 0..63
  const int wave = tid >> 5;                  // oc tile
  const int lane = tid & 31;
  const int tile = blockIdx.x;
  const int n    = blockIdx.y >> 2;
  const int g    = blockIdx.y & 3;

  // ---- staging role: thread t gathers pixel (t&15), ci block (t>>4)*16
  const int sm  = tid & 15;                   // pixel-in-tile (coalesced dim)
  const int sc0 = (tid >> 4) * 16;            // ci base
  const int spa = tile * 16 + sm;
  const int soy = spa / W, sox = spa % W;
  const bool sok = spa < HW;
  const float* xin = mid + ((size_t)(n * CC + g * CGc)) * HW;

  // ---- compute role (per wave)
  const int  mrow  = lane & 15;               // A row / D col index
  const bool hif   = (lane & 16) != 0;
  const int  aKofs = hif ? 8 : 0;             // A lanes 16-31 carry K 8-15/24-31
  const int  bKofs = hif ? 8 : 0;             // B lanes 16-31 carry K 8-15/24-31
  const int  oc    = wave * 16 + mrow;        // padded out channel (0..31)
  const bool oc_ok = oc < OCG;
  const _Float16* wB = wf + ((size_t)(g * OCP + oc)) * KTOT;  // K-contiguous
  __builtin_prefetch(wB, 0, 0);               // CDNA5 global_prefetch_b8

  v8f acc = {};
#pragma unroll 3
  for (int tap = 0; tap < 9; ++tap) {         // one 64-K super-chunk per tap
    // ---- stage A super-chunk: fixed tap -> single bounds test per thread
    {
      int iy = soy + (tap / 3) - 1, ix = sox + (tap % 3) - 1;
      const float* src = (sok && iy >= 0 && iy < H && ix >= 0 && ix < W)
                             ? (xin + (size_t)iy * W + ix) : nullptr;
#pragma unroll
      for (int e = 0; e < 16; ++e) {
        int ci = sc0 + e;
        float v = src ? src[(size_t)ci * HW] : 0.f;
        Alds[sm * 64 + ci] = (_Float16)v;
      }
    }
    __syncthreads();

    // ---- two 32-K WMMA chunks out of the super-chunk
    const _Float16* arow = &Alds[mrow * 64];
    const _Float16* brow = wB + tap * 64;
#pragma unroll
    for (int half = 0; half < 2; ++half) {
      const int co = half * 32;
      v8h alo = *(const v8h*)(arow + co + aKofs);        // ds_load_b128
      v8h ahi = *(const v8h*)(arow + co + aKofs + 16);   // ds_load_b128
      v8h blo = *(const v8h*)(brow + co + bKofs);        // global_load_b128
      v8h bhi = *(const v8h*)(brow + co + bKofs + 16);   // global_load_b128
      v16h A, B;
#pragma unroll
      for (int e = 0; e < 8; ++e) {
        A[e] = alo[e]; A[e + 8] = ahi[e];
        B[e] = blo[e]; B[e + 8] = bhi[e];
      }
      acc = __builtin_amdgcn_wmma_f32_16x16x32_f16(false, A, false, B,
                                                   (short)0, acc, false, false);
    }
    __syncthreads();
  }

  float bias = oc_ok ? ob[g * OCG + oc] : 0.f;
#pragma unroll
  for (int v = 0; v < 8; ++v) {
    int pd = tile * 16 + v + (hif ? 8 : 0);   // D: M = vgpr + 8*hi, N = lane&15
    if (pd < HW && oc_ok)
      om[((size_t)n * OMC + g * OCG + oc) * HW + pd] = acc[v] + bias;
  }
}

// ---------------- DCNv3 bilinear sampling -----------------------------------
__global__ void dcnv3_kernel(const float* __restrict__ x, const float* __restrict__ om,
                             float* __restrict__ out, int H, int W, int Ho, int Wo,
                             int omH, int omW, int stride, int softmax) {
  int idx = blockIdx.x * blockDim.x + threadIdx.x;
  int total = NB * CC * Ho * Wo;
  if (idx >= total) return;
  int p = idx % (Ho * Wo);
  int c = (idx / (Ho * Wo)) % CC;
  int n = idx / (CC * Ho * Wo);
  int oy = p / Wo, ox = p % Wo;
  int g = c >> 6;
  int omy = oy * stride, omx = ox * stride;
  size_t ombase = ((size_t)n * OMC) * omH * omW + (size_t)omy * omW + omx;
  size_t plane  = (size_t)omH * omW;

  float offy[9], offx[9], mk[9];
#pragma unroll
  for (int k = 0; k < 9; ++k) {
    offy[k] = om[ombase + (size_t)(g * 18 + 2 * k + 0) * plane];
    offx[k] = om[ombase + (size_t)(g * 18 + 2 * k + 1) * plane];
    mk[k]   = om[ombase + (size_t)(72 + g * 9 + k) * plane];
  }
  if (softmax) {
    float mx = mk[0];
#pragma unroll
    for (int k = 1; k < 9; ++k) mx = fmaxf(mx, mk[k]);
    float sm = 0.f;
#pragma unroll
    for (int k = 0; k < 9; ++k) { mk[k] = expf(mk[k] - mx); sm += mk[k]; }
    float inv = 1.f / sm;
#pragma unroll
    for (int k = 0; k < 9; ++k) mk[k] *= inv;
  } else {
#pragma unroll
    for (int k = 0; k < 9; ++k) mk[k] = sigf(mk[k]);
  }

  const float* xp = x + ((size_t)(n * CC + c)) * H * W;
  float acc = 0.f;
#pragma unroll
  for (int k = 0; k < 9; ++k) {
    float py = (float)(oy * stride + (k / 3) - 1) + offy[k];
    float px = (float)(ox * stride + (k % 3) - 1) + offx[k];
    float y0f = floorf(py), x0f = floorf(px);
    float wy = py - y0f, wx = px - x0f;
    int y0 = (int)y0f, x0 = (int)x0f;
#pragma unroll
    for (int dy = 0; dy < 2; ++dy)
#pragma unroll
      for (int dx = 0; dx < 2; ++dx) {
        int yi = y0 + dy, xi = x0 + dx;
        if (yi >= 0 && yi < H && xi >= 0 && xi < W) {
          float wgt = (dy ? wy : 1.f - wy) * (dx ? wx : 1.f - wx) * mk[k];
          acc += wgt * xp[(size_t)yi * W + xi];
        }
      }
  }
  out[idx] = acc;
}

// ---------------- GroupNorm (16 groups of 16 channels) ----------------------
__global__ void group_norm_kernel(const float* __restrict__ x, const float* __restrict__ gamma,
                                  const float* __restrict__ beta, float* __restrict__ y,
                                  int H, int W) {
  int n = blockIdx.x >> 4, grp = blockIdx.x & 15;
  int tid = threadIdx.x;
  int HW = H * W;
  int sz = 16 * HW;
  const float* xp = x + ((size_t)(n * CC + grp * 16)) * HW;
  float*       yp = y + ((size_t)(n * CC + grp * 16)) * HW;
  float s = 0.f, ss = 0.f;
  for (int i = tid; i < sz; i += 256) { float v = xp[i]; s += v; ss += v * v; }
  __shared__ float rs[256], rq[256];
  rs[tid] = s; rq[tid] = ss;
  __syncthreads();
  for (int st = 128; st > 0; st >>= 1) {
    if (tid < st) { rs[tid] += rs[tid + st]; rq[tid] += rq[tid + st]; }
    __syncthreads();
  }
  float mu  = rs[0] / (float)sz;
  float var = rq[0] / (float)sz - mu * mu;
  float inv = rsqrtf(var + 1e-5f);
  for (int i = tid; i < sz; i += 256) {
    int c = grp * 16 + i / HW;
    yp[i] = (xp[i] - mu) * inv * gamma[c] + beta[c];
  }
}

// ---------------- global average pool (per n,c) -----------------------------
__global__ void pool_kernel(const float* __restrict__ x, float* __restrict__ pooled,
                            int HW, float scale) {
  int nc = blockIdx.x, tid = threadIdx.x;
  const float* xp = x + (size_t)nc * HW;
  float s = 0.f;
  for (int i = tid; i < HW; i += 256) s += xp[i];
  __shared__ float rs[256];
  rs[tid] = s;
  __syncthreads();
  for (int st = 128; st > 0; st >>= 1) {
    if (tid < st) rs[tid] += rs[tid + st];
    __syncthreads();
  }
  if (tid == 0) pooled[nc] = rs[0] / (float)HW * scale;
}

// ---------------- scale-attention scalar: hardsigmoid(relu(y.w + b)) --------
__global__ void attn_kernel(const float* __restrict__ pooled, const float* __restrict__ saw,
                            const float* __restrict__ sab, float* __restrict__ attn) {
  int n = blockIdx.x, t = threadIdx.x;
  __shared__ float rs[256];
  rs[t] = pooled[n * CC + t] * saw[t];
  __syncthreads();
  for (int st = 128; st > 0; st >>= 1) {
    if (t < st) rs[t] += rs[t + st];
    __syncthreads();
  }
  if (t == 0) {
    float a = fmaxf(rs[0] + sab[0], 0.f);
    attn[n] = fminf(fmaxf((a + 3.f) / 6.f, 0.f), 1.f);
  }
}

// ---------------- sum += feat * attn[n] -------------------------------------
__global__ void accum_kernel(const float* __restrict__ feat, const float* __restrict__ attn,
                             float* __restrict__ sum, int HW, int first) {
  int idx = blockIdx.x * blockDim.x + threadIdx.x;
  int total = NB * CC * HW;
  if (idx >= total) return;
  int n = idx / (CC * HW);
  float v = feat[idx] * attn[n];
  sum[idx] = first ? v : sum[idx] + v;
}

// ---------------- bilinear upsample, align-corners --------------------------
__global__ void upsample_kernel(const float* __restrict__ x, float* __restrict__ y,
                                int H, int W, int Ho, int Wo) {
  int idx = blockIdx.x * blockDim.x + threadIdx.x;
  int total = NB * CC * Ho * Wo;
  if (idx >= total) return;
  int ox = idx % Wo, oy = (idx / Wo) % Ho;
  int nc = idx / (Ho * Wo);
  float sy = (Ho > 1) ? (float)(H - 1) / (float)(Ho - 1) : 0.f;
  float sx = (Wo > 1) ? (float)(W - 1) / (float)(Wo - 1) : 0.f;
  float cy = oy * sy, cx = ox * sx;
  int y0 = (int)floorf(cy), x0 = (int)floorf(cx);
  int y1 = min(y0 + 1, H - 1), x1 = min(x0 + 1, W - 1);
  float wy = cy - (float)y0, wx = cx - (float)x0;
  const float* xp = x + (size_t)nc * H * W;
  float t0 = xp[(size_t)y0 * W + x0] * (1.f - wy) + xp[(size_t)y1 * W + x0] * wy;
  float t1 = xp[(size_t)y0 * W + x1] * (1.f - wy) + xp[(size_t)y1 * W + x1] * wy;
  y[idx] = t0 * (1.f - wx) + t1 * wx;
}

// ---------------- DyReLU MLP (tiny, N=4 rows) -------------------------------
__global__ void fc1_kernel(const float* __restrict__ pooled, const float* __restrict__ w,
                           const float* __restrict__ b, float* __restrict__ h) {
  int n = blockIdx.x, j = threadIdx.x;  // 64 threads
  float s = b[j];
  for (int c = 0; c < CC; ++c) s += pooled[n * CC + c] * w[j * CC + c];
  h[n * 64 + j] = fmaxf(s, 0.f);
}

__global__ void fc2_kernel(const float* __restrict__ h, const float* __restrict__ w,
                           const float* __restrict__ b, float* __restrict__ z) {
  int n = blockIdx.x, j = threadIdx.x;  // 1024 threads
  float s = b[j];
  for (int c = 0; c < 64; ++c) s += h[n * 64 + c] * w[j * 64 + c];
  z[n * 1024 + j] = fminf(fmaxf(s + 3.f, 0.f), 6.f) / 6.f;
}

__global__ void dyrelu_kernel(const float* __restrict__ sum, const float* __restrict__ z,
                              float* __restrict__ out, int HW, float inv_nl) {
  int idx = blockIdx.x * blockDim.x + threadIdx.x;
  int total = NB * CC * HW;
  if (idx >= total) return;
  int n = idx / (CC * HW);
  int c = (idx / HW) % CC;
  float f = sum[idx] * inv_nl;
  const float* zn = z + n * 1024;
  float a1 = (zn[c]       - 0.5f) * 2.f + 1.f;
  float b1 =  zn[256 + c] - 0.5f;
  float a2 = (zn[512 + c] - 0.5f) * 2.f;
  float b2 =  zn[768 + c] - 0.5f;
  out[idx] = fmaxf(f * a1 + b1, f * a2 + b2);
}

// ---------------------------------------------------------------------------
extern "C" void kernel_launch(void* const* d_in, const int* in_sizes, int n_in,
                              void* d_out, int out_size, void* d_ws, size_t ws_size,
                              hipStream_t stream) {
  (void)in_sizes; (void)n_in; (void)out_size; (void)ws_size;
  const float* xs[3]  = {(const float*)d_in[0], (const float*)d_in[1], (const float*)d_in[2]};
  const float* dw_w_h = (const float*)d_in[3],  *dw_s_h = (const float*)d_in[4],  *dw_b_h = (const float*)d_in[5];
  const float* dw_w_m = (const float*)d_in[6],  *dw_s_m = (const float*)d_in[7],  *dw_b_m = (const float*)d_in[8];
  const float* dw_w_l = (const float*)d_in[9],  *dw_s_l = (const float*)d_in[10], *dw_b_l = (const float*)d_in[11];
  const float* off_w  = (const float*)d_in[12], *off_b  = (const float*)d_in[13];
  const float* gn_g_h = (const float*)d_in[14], *gn_b_h = (const float*)d_in[15];
  const float* gn_g_m = (const float*)d_in[16], *gn_b_m = (const float*)d_in[17];
  const float* gn_g_l = (const float*)d_in[18], *gn_b_l = (const float*)d_in[19];
  const float* sa_w   = (const float*)d_in[20], *sa_b   = (const float*)d_in[21];
  const float* fc1_w  = (const float*)d_in[22], *fc1_b  = (const float*)d_in[23];
  const float* fc2_w  = (const float*)d_in[24], *fc2_b  = (const float*)d_in[25];
  float* out = (float*)d_out;

  // ---- workspace carving (floats; all segments 16B-aligned) ----
  const size_t BMAX = (size_t)NB * CC * 80 * 80;        // 6,553,600
  const size_t OMAX = (size_t)NB * OMC * 80 * 80;       // 2,764,800
  float* ws       = (float*)d_ws;
  float* buf_dw   = ws;
  float* buf_om   = buf_dw   + BMAX;
  float* buf_samp = buf_om   + OMAX;
  float* buf_feat = buf_samp + BMAX;
  float* buf_up   = buf_feat + BMAX;
  float* buf_sum  = buf_up   + BMAX;
  float* pooled   = buf_sum  + BMAX;    // NB*CC
  float* attn     = pooled   + NB * CC; // 8 (padded)
  float* hb       = attn     + 8;       // NB*64
  float* zb       = hb       + NB * 64; // NB*1024
  _Float16* wf16  = (_Float16*)(zb + NB * 1024);  // GG*OCP*KTOT f16

  auto cdiv = [](int a, int b) { return (a + b - 1) / b; };
  const int Hs[3] = {80, 40, 20};

  // pack offset-conv weights once per launch (deterministic, capture-safe)
  pack_weights_kernel<<<cdiv(GG * OCP * KTOT, 256), 256, 0, stream>>>(off_w, wf16);

  // one dw->offsetconv->dcnv3->gn path
  auto run_path = [&](const float* x, int Hin,
                      const float* dww, const float* dws, const float* dwb,
                      const float* gng, const float* gnb,
                      int stride, int softmax, float* featbuf, int Ho) {
    int HWin = Hin * Hin;
    int tin  = NB * CC * HWin;
    dw_silu_kernel<<<cdiv(tin, 256), 256, 0, stream>>>(x, dww, dws, dwb, buf_dw, Hin, Hin);
    dim3 og(cdiv(HWin, 16), NB * GG);
    offset_conv_wmma_kernel<<<og, 64, 0, stream>>>(buf_dw, wf16, off_b, buf_om, Hin, Hin);
    int tout = NB * CC * Ho * Ho;
    dcnv3_kernel<<<cdiv(tout, 256), 256, 0, stream>>>(x, buf_om, buf_samp,
                                                      Hin, Hin, Ho, Ho, Hin, Hin,
                                                      stride, softmax);
    group_norm_kernel<<<NB * 16, 256, 0, stream>>>(buf_samp, gng, gnb, featbuf, Ho, Ho);
  };

  auto scale_accum = [&](const float* feat, int H, int first) {
    int HW = H * H;
    pool_kernel<<<NB * CC, 256, 0, stream>>>(feat, pooled, HW, 1.f);
    attn_kernel<<<NB, 256, 0, stream>>>(pooled, sa_w, sa_b, attn);
    accum_kernel<<<cdiv(NB * CC * HW, 256), 256, 0, stream>>>(feat, attn, buf_sum, HW, first);
  };

  size_t out_off = 0;
  for (int lvl = 0; lvl < 3; ++lvl) {
    int H = Hs[lvl], HW = H * H;
    int nl = 1;

    // MID path (sigmoid mask, stride 1)
    run_path(xs[lvl], H, dw_w_m, dw_s_m, dw_b_m, gn_g_m, gn_b_m,
             /*stride=*/1, /*softmax=*/0, buf_feat, H);
    scale_accum(buf_feat, H, /*first=*/1);

    // LOW path (from finer level, stride-2 sampling, softmax mask)
    if (lvl > 0) {
      int Hl = Hs[lvl - 1];  // = 2H
      run_path(xs[lvl - 1], Hl, dw_w_l, dw_s_l, dw_b_l, gn_g_l, gn_b_l,
               /*stride=*/2, /*softmax=*/1, buf_feat, H);
      scale_accum(buf_feat, H, 0);
      ++nl;
    }

    // HIGH path (from coarser level, upsampled, softmax mask)
    if (lvl < 2) {
      int Hh = Hs[lvl + 1];  // = H/2
      run_path(xs[lvl + 1], Hh, dw_w_h, dw_s_h, dw_b_h, gn_g_h, gn_b_h,
               /*stride=*/1, /*softmax=*/1, buf_feat, Hh);
      int tup = NB * CC * HW;
      upsample_kernel<<<cdiv(tup, 256), 256, 0, stream>>>(buf_feat, buf_up, Hh, Hh, H, H);
      scale_accum(buf_up, H, 0);
      ++nl;
    }

    // DyReLU head on sum/nl
    float inv_nl = 1.f / (float)nl;
    pool_kernel<<<NB * CC, 256, 0, stream>>>(buf_sum, pooled, HW, inv_nl);
    fc1_kernel<<<NB, 64, 0, stream>>>(pooled, fc1_w, fc1_b, hb);
    fc2_kernel<<<NB, 1024, 0, stream>>>(hb, fc2_w, fc2_b, zb);
    dyrelu_kernel<<<cdiv(NB * CC * HW, 256), 256, 0, stream>>>(buf_sum, zb, out + out_off,
                                                               HW, inv_nl);
    out_off += (size_t)NB * CC * HW;
  }
}
